// PSR_GNN_Branch_34926674051129
// MI455X (gfx1250) — matrix operand order repaired and checked by reading, not compiled, not run on hardware
//
#include <hip/hip_runtime.h>
#include <hip/hip_bf16.h>

// ---- problem constants (match reference) ----
#define BB 8
#define LL 64
#define MMM 48
#define DD 16
#define HID 256
#define OUTC 512
#define HEADS 4
#define KNN 3
#define PREDL 96
#define ENCIN 32
#define N_PER (LL * MMM)      // 3072
#define NTOT (BB * N_PER)     // 24576
#define BN_EPS 1e-5f

typedef _Float16 half16 __attribute__((ext_vector_type(16)));
typedef float    float8 __attribute__((ext_vector_type(8)));

// ============================================================
// kNN: one block per node, brute-force over its graph (3072 pts)
// ============================================================
__global__ void __launch_bounds__(128) knn_kernel(const float* __restrict__ x,
                                                  int* __restrict__ src)
{
  const int i    = blockIdx.x;
  const int g    = i / N_PER;
  const int il   = i - g * N_PER;
  const int base = g * N_PER;
  const int t    = threadIdx.x;

  __shared__ float xi[DD];
  __shared__ float bd[128 * KNN];
  __shared__ int   bi[128 * KNN];

  if (t < DD) xi[t] = x[(size_t)i * DD + t];
  __syncthreads();

  float best[KNN]; int bidx[KNN];
  #pragma unroll
  for (int k = 0; k < KNN; ++k) { best[k] = INFINITY; bidx[k] = 0x7fffffff; }

  for (int j = t; j < N_PER; j += 128) {
    if (j == il) continue;
    const float* xj = &x[(size_t)(base + j) * DD];
    float d2 = 0.f;
    #pragma unroll
    for (int d = 0; d < DD; ++d) { float df = xi[d] - xj[d]; d2 += df * df; }
    #pragma unroll
    for (int k = 0; k < KNN; ++k) {
      if (d2 < best[k] || (d2 == best[k] && j < bidx[k])) {
        for (int m = KNN - 1; m > k; --m) { best[m] = best[m-1]; bidx[m] = bidx[m-1]; }
        best[k] = d2; bidx[k] = j;
        break;
      }
    }
  }
  #pragma unroll
  for (int k = 0; k < KNN; ++k) { bd[t*KNN + k] = best[k]; bi[t*KNN + k] = bidx[k]; }
  __syncthreads();

  if (t == 0) {  // serial merge of 384 candidates -> top-3 (tiny)
    for (int k = 0; k < KNN; ++k) {
      float bm = INFINITY; int im = 0x7fffffff; int pos = -1;
      for (int q = 0; q < 128 * KNN; ++q) {
        float d = bd[q]; int jj = bi[q];
        if (d < bm || (d == bm && jj < im)) { bm = d; im = jj; pos = q; }
      }
      if (pos >= 0) bd[pos] = INFINITY;
      src[(size_t)i * KNN + k] = base + im;
    }
  }
}

// ============================================================
// GATConv, fully node-local: each target has exactly 4 incoming
// edges (3 kNN sources + self). Recompute the 4 needed h rows
// (x[row] @ gat_w, K=16) in LDS instead of materializing 100MB h.
// ============================================================
__global__ void __launch_bounds__(256) gat_kernel(
    const float* __restrict__ x,    const float* __restrict__ gw,
    const float* __restrict__ asrc, const float* __restrict__ adst,
    const float* __restrict__ gbias,const int*   __restrict__ src,
    float* __restrict__ xgat)
{
  const int i = blockIdx.x;
  const int t = threadIdx.x;

  __shared__ int   sidx[4];
  __shared__ float xs[4 * DD];
  __shared__ float hrows[4 * HEADS * HID];   // 16 KB
  __shared__ float red[256];
  __shared__ float asv[4][HEADS];
  __shared__ float adv[HEADS];
  __shared__ float alpha[4][HEADS];

  if (t < KNN)  sidx[t]   = src[(size_t)i * KNN + t];
  if (t == KNN) sidx[KNN] = i;                 // self loop row
  __syncthreads();
  if (t < 4 * DD) xs[t] = x[(size_t)sidx[t >> 4] * DD + (t & 15)];
  __syncthreads();

  // h rows for {j0,j1,j2,self}: 4 x 1024 outputs, 16 MACs each
  for (int o = t; o < 4 * 1024; o += 256) {
    int row = o >> 10, col = o & 1023;
    float s = 0.f;
    #pragma unroll
    for (int d = 0; d < DD; ++d) s += xs[row * DD + d] * gw[d * 1024 + col];
    hrows[o] = s;
  }
  __syncthreads();

  // a_s for all 4 rows, a_d for self row: 20 block reductions of 256
  for (int rh = 0; rh < 20; ++rh) {
    int row, hd; const float* av;
    if (rh < 16) { row = rh >> 2; hd = rh & 3; av = asrc; }
    else         { row = 3;       hd = rh - 16; av = adst; }
    red[t] = hrows[row * 1024 + hd * HID + t] * av[hd * HID + t];
    __syncthreads();
    for (int s = 128; s > 0; s >>= 1) {
      if (t < s) red[t] += red[t + s];
      __syncthreads();
    }
    if (t == 0) { if (rh < 16) asv[row][hd] = red[0]; else adv[hd] = red[0]; }
    __syncthreads();
  }

  // per-head softmax over the 4 edges (leaky_relu 0.2)
  if (t < HEADS) {
    int hd = t;
    float e[4], m = -INFINITY;
    #pragma unroll
    for (int ed = 0; ed < 4; ++ed) {
      float v = asv[ed][hd] + adv[hd];
      v = (v > 0.f) ? v : 0.2f * v;
      e[ed] = v; m = fmaxf(m, v);
    }
    float den = 0.f;
    #pragma unroll
    for (int ed = 0; ed < 4; ++ed) { e[ed] = __expf(e[ed] - m); den += e[ed]; }
    den += 1e-16f;
    #pragma unroll
    for (int ed = 0; ed < 4; ++ed) alpha[ed][hd] = e[ed] / den;
  }
  __syncthreads();

  // aggregate, head-mean, bias
  {
    const int c = t;
    float tot = 0.f;
    #pragma unroll
    for (int hd = 0; hd < HEADS; ++hd) {
      float s = 0.f;
      #pragma unroll
      for (int ed = 0; ed < 4; ++ed) s += alpha[ed][hd] * hrows[ed * 1024 + hd * HID + c];
      tot += s;
    }
    xgat[(size_t)i * HID + c] = tot * 0.25f + gbias[c];
  }
}

// ============================================================
// BatchNorm (training mode, biased var) — column stats + apply
// ============================================================
__global__ void __launch_bounds__(256) bn_stats_kernel(const float* __restrict__ x,
                                                       float* __restrict__ mean,
                                                       float* __restrict__ rstd,
                                                       int rows, int cols)
{
  const int c = blockIdx.x;
  const int t = threadIdx.x;
  __shared__ float s1[256], s2[256];
  float a = 0.f, b = 0.f;
  for (int r = t; r < rows; r += 256) {
    float v = x[(size_t)r * cols + c];
    a += v; b += v * v;
  }
  s1[t] = a; s2[t] = b; __syncthreads();
  for (int s = 128; s > 0; s >>= 1) {
    if (t < s) { s1[t] += s1[t + s]; s2[t] += s2[t + s]; }
    __syncthreads();
  }
  if (t == 0) {
    float mu  = s1[0] / rows;
    float var = s2[0] / rows - mu * mu;
    mean[c] = mu;
    rstd[c] = rsqrtf(var + BN_EPS);
  }
}

__global__ void bn_apply_relu_kernel(float* __restrict__ x,
                                     const float* __restrict__ mean,
                                     const float* __restrict__ rstd,
                                     const float* __restrict__ g,
                                     const float* __restrict__ b,
                                     int rows, int cols)
{
  size_t idx = (size_t)blockIdx.x * blockDim.x + threadIdx.x;
  size_t tot = (size_t)rows * cols;
  if (idx >= tot) return;
  int c = (int)(idx % cols);
  float v = (x[idx] - mean[c]) * rstd[c] * g[c] + b[c];
  x[idx] = fmaxf(v, 0.f);
}

// ============================================================
// GIN pre: (1+eps)*x[i] + sum_{j in kNN(i)} x[j].
// Neighbor rows (3 x 256 contiguous floats) are pulled into LDS
// with GLOBAL_LOAD_ASYNC_TO_LDS_B32 (ASYNCcnt path, §15.18.3
// op 96; VDST = per-lane LDS byte offset, VADDR = 64-bit global
// address), completed with s_wait_asynccnt + barrier.
// ============================================================
__global__ void __launch_bounds__(256) gin_pre_kernel(const float* __restrict__ xgat,
                                                      const int* __restrict__ src,
                                                      const float* __restrict__ eps,
                                                      float* __restrict__ out)
{
  const int i = blockIdx.x;
  const int c = threadIdx.x;
  __shared__ float nb0[HID], nb1[HID], nb2[HID];

  int j0 = src[i * 3 + 0], j1 = src[i * 3 + 1], j2 = src[i * 3 + 2];

  // per-lane LDS byte offsets (generic shared ptr truncates to LDS offset)
  unsigned l0 = (unsigned)(unsigned long)(const void*)&nb0[c];
  unsigned l1 = (unsigned)(unsigned long)(const void*)&nb1[c];
  unsigned l2 = (unsigned)(unsigned long)(const void*)&nb2[c];
  const float* g0 = xgat + (size_t)j0 * HID + c;
  const float* g1 = xgat + (size_t)j1 * HID + c;
  const float* g2 = xgat + (size_t)j2 * HID + c;

  asm volatile("global_load_async_to_lds_b32 %0, %1, off" :: "v"(l0), "v"(g0) : "memory");
  asm volatile("global_load_async_to_lds_b32 %0, %1, off" :: "v"(l1), "v"(g1) : "memory");
  asm volatile("global_load_async_to_lds_b32 %0, %1, off" :: "v"(l2), "v"(g2) : "memory");
  asm volatile("s_wait_asynccnt 0x0" ::: "memory");
  __syncthreads();

  out[(size_t)i * HID + c] =
      (1.f + eps[0]) * xgat[(size_t)i * HID + c] + nb0[c] + nb1[c] + nb2[c];
}

// ============================================================
// Weight transpose + f16 convert: Bt[N][K] = (f16)B[K][N].
// Makes the WMMA B-fragment (K = s + hi*16, contiguous in K)
// a single aligned 32-byte load per lane.
// ============================================================
__global__ void __launch_bounds__(256) wtrans_kernel(const float* __restrict__ B,
                                                     _Float16* __restrict__ Bt,
                                                     int K, int N)
{
  long tid = (long)blockIdx.x * 256 + threadIdx.x;
  if (tid >= (long)K * N) return;
  int k = (int)(tid / N), c = (int)(tid % N);
  Bt[(size_t)c * K + k] = (_Float16)B[tid];
}

// ============================================================
// GEMM on v_wmma_f32_16x16x32_f16 with 16x64 register blocking:
// each wave owns a 16(M) x 64(N) output tile = 4 accumulators.
// Per K-step: one A fragment (4x b128 + 8 pk-cvt) is reused by
// four B fragments (8x b128) -> 4 WMMAs. A: fp32 [M x K] row-major,
// Bt: f16 [N x K] pre-transposed. K mult of 32, N mult of 64,
// A rows padded by caller -> no guards in the hot loop.
// Fragment layouts used here (wave32; contiguous K, same order in A and B):
//   A lane l slot s -> row = l&15, K = k0 + hi*16 + s
//   B lane l slot s -> col = l&15, K = k0 + hi*16 + s (contig 32B)
//   C/D lane l elem v -> row = v + hi*8, col = l&15
// ============================================================
__global__ void __launch_bounds__(128) gemm_wmma_kernel(
    const float* __restrict__ A, const _Float16* __restrict__ Bt,
    const float* __restrict__ bias, const float* __restrict__ residual,
    float* __restrict__ D, int Mstore, int Ncols, int Kc, int relu,
    int ntilesN, int ntilesTot)   // ntilesN counts 64-wide N tiles
{
  const int wave = threadIdx.x >> 5;
  const int lane = threadIdx.x & 31;
  const int tile = blockIdx.x * 4 + wave;
  if (tile >= ntilesTot) return;          // wave-uniform exit
  const int mt = tile / ntilesN;
  const int nt = tile - mt * ntilesN;
  const int hi  = lane >> 4;
  const int l15 = lane & 15;

  const float*    arow = A  + (size_t)(mt * 16 + l15) * Kc;
  const _Float16* bc0  = Bt + (size_t)(nt * 64 +      l15) * Kc;
  const _Float16* bc1  = Bt + (size_t)(nt * 64 + 16 + l15) * Kc;
  const _Float16* bc2  = Bt + (size_t)(nt * 64 + 32 + l15) * Kc;
  const _Float16* bc3  = Bt + (size_t)(nt * 64 + 48 + l15) * Kc;

  float8 acc0 = {}, acc1 = {}, acc2 = {}, acc3 = {};
  for (int k0 = 0; k0 < Kc; k0 += 32) {
    const float* pa = arow + k0 + hi * 16;
    if (k0 + 32 < Kc) __builtin_prefetch(pa + 32, 0, 3);   // global_prefetch_b8
    float4 a0 = *(const float4*)(pa);
    float4 a1 = *(const float4*)(pa + 4);
    float4 a2 = *(const float4*)(pa + 8);
    float4 a3 = *(const float4*)(pa + 12);
    half16 af;
    af[0]=(_Float16)a0.x;  af[1]=(_Float16)a0.y;  af[2]=(_Float16)a0.z;  af[3]=(_Float16)a0.w;
    af[4]=(_Float16)a1.x;  af[5]=(_Float16)a1.y;  af[6]=(_Float16)a1.z;  af[7]=(_Float16)a1.w;
    af[8]=(_Float16)a2.x;  af[9]=(_Float16)a2.y;  af[10]=(_Float16)a2.z; af[11]=(_Float16)a2.w;
    af[12]=(_Float16)a3.x; af[13]=(_Float16)a3.y; af[14]=(_Float16)a3.z; af[15]=(_Float16)a3.w;

    const int kb = k0 + hi * 16;
    half16 bf0 = *(const half16*)(bc0 + kb);   // 32B aligned each
    half16 bf1 = *(const half16*)(bc1 + kb);
    half16 bf2 = *(const half16*)(bc2 + kb);
    half16 bf3 = *(const half16*)(bc3 + kb);

    acc0 = __builtin_amdgcn_wmma_f32_16x16x32_f16(false, af, false, bf0, (short)0, acc0, false, false);
    acc1 = __builtin_amdgcn_wmma_f32_16x16x32_f16(false, af, false, bf1, (short)0, acc1, false, false);
    acc2 = __builtin_amdgcn_wmma_f32_16x16x32_f16(false, af, false, bf2, (short)0, acc2, false, false);
    acc3 = __builtin_amdgcn_wmma_f32_16x16x32_f16(false, af, false, bf3, (short)0, acc3, false, false);
  }

  // epilogue: 4 sub-tiles of 16 columns each
  #pragma unroll
  for (int q = 0; q < 4; ++q) {
    const float8& acc = (q == 0) ? acc0 : (q == 1) ? acc1 : (q == 2) ? acc2 : acc3;
    const int col = nt * 64 + q * 16 + l15;
    #pragma unroll
    for (int v = 0; v < 8; ++v) {
      int row = mt * 16 + v + hi * 8;
      if (row < Mstore) {
        float val = acc[v];
        if (bias)     val += bias[col];
        if (residual) val += residual[(size_t)row * Ncols + col];
        if (relu)     val = fmaxf(val, 0.f);
        D[(size_t)row * Ncols + col] = val;
      }
    }
  }
}

// ============================================================
// global mean pool; writes a 16-row buffer (rows 8..15 zeroed
// so the final WMMA GEMM can read a full padded 16-row A tile)
// ============================================================
__global__ void __launch_bounds__(256) pool_kernel(const float* __restrict__ node,
                                                   float* __restrict__ graph)
{
  const int b = blockIdx.x;                      // 0..15 (padded)
  const int c = blockIdx.y * 256 + threadIdx.x;
  float s = 0.f;
  if (b < BB) {
    for (int n = 0; n < N_PER; ++n) s += node[((size_t)b * N_PER + n) * OUTC + c];
    s *= (1.f / N_PER);
  }
  graph[(size_t)b * OUTC + c] = s;
}

// ============================================================
extern "C" void kernel_launch(void* const* d_in, const int* in_sizes, int n_in,
                              void* d_out, int out_size, void* d_ws, size_t ws_size,
                              hipStream_t stream)
{
  (void)in_sizes; (void)n_in; (void)out_size; (void)ws_size;

  const float* x_chaos  = (const float*)d_in[0];
  const float* gat_w    = (const float*)d_in[1];
  const float* att_src  = (const float*)d_in[2];
  const float* att_dst  = (const float*)d_in[3];
  const float* gat_bias = (const float*)d_in[4];
  const float* gat_bn_g = (const float*)d_in[5];
  const float* gat_bn_b = (const float*)d_in[6];
  const float* gin_eps  = (const float*)d_in[7];
  const float* gin_w1   = (const float*)d_in[8];
  const float* gin_b1   = (const float*)d_in[9];
  const float* gin_bn_g = (const float*)d_in[10];
  const float* gin_bn_b = (const float*)d_in[11];
  const float* gin_w2   = (const float*)d_in[12];
  const float* gin_b2   = (const float*)d_in[13];
  const float* res_w    = (const float*)d_in[14];
  const float* res_b    = (const float*)d_in[15];
  const float* fin_w    = (const float*)d_in[16];
  const float* fin_b    = (const float*)d_in[17];

  char* ws = (char*)d_ws;
  size_t off = 0;
  auto alloc = [&](size_t bytes) -> void* {
    void* p = ws + off;
    off = (off + bytes + 255) & ~(size_t)255;
    return p;
  };
  int*       knn   = (int*)      alloc((size_t)NTOT * KNN * sizeof(int));
  float*     xgat  = (float*)    alloc((size_t)NTOT * HID  * sizeof(float));  // 25 MB
  float*     bufA  = (float*)    alloc((size_t)NTOT * HID  * sizeof(float));  // 25 MB
  float*     bufB  = (float*)    alloc((size_t)NTOT * HID  * sizeof(float));  // 25 MB
  float*     bufC  = (float*)    alloc((size_t)NTOT * OUTC * sizeof(float));  // 50 MB
  _Float16*  w1t   = (_Float16*) alloc((size_t)HID  * HID  * sizeof(_Float16));
  _Float16*  w2t   = (_Float16*) alloc((size_t)OUTC * HID  * sizeof(_Float16));
  _Float16*  rwt   = (_Float16*) alloc((size_t)OUTC * HID  * sizeof(_Float16));
  _Float16*  fwt   = (_Float16*) alloc((size_t)(PREDL*ENCIN) * OUTC * sizeof(_Float16));
  float*     mean1 = (float*)    alloc(HID * sizeof(float));
  float*     rstd1 = (float*)    alloc(HID * sizeof(float));
  float*     mean2 = (float*)    alloc(HID * sizeof(float));
  float*     rstd2 = (float*)    alloc(HID * sizeof(float));
  float*     graph = (float*)    alloc((size_t)16 * OUTC * sizeof(float));    // padded to 16 rows

  // 0. weight transpose+f16 (independent of data path)
  wtrans_kernel<<<(HID*HID + 255)/256, 256, 0, stream>>>(gin_w1, w1t, HID, HID);
  wtrans_kernel<<<(HID*OUTC + 255)/256, 256, 0, stream>>>(gin_w2, w2t, HID, OUTC);
  wtrans_kernel<<<(HID*OUTC + 255)/256, 256, 0, stream>>>(res_w,  rwt, HID, OUTC);
  wtrans_kernel<<<(OUTC*PREDL*ENCIN + 255)/256, 256, 0, stream>>>(fin_w, fwt, OUTC, PREDL*ENCIN);

  // 1. kNN edges
  knn_kernel<<<NTOT, 128, 0, stream>>>(x_chaos, knn);
  // 2. GATConv (fused h recompute + 4-edge softmax + head mean + bias)
  gat_kernel<<<NTOT, 256, 0, stream>>>(x_chaos, gat_w, att_src, att_dst, gat_bias, knn, xgat);
  // 3. BN + ReLU (in place)
  bn_stats_kernel<<<HID, 256, 0, stream>>>(xgat, mean1, rstd1, NTOT, HID);
  bn_apply_relu_kernel<<<(NTOT*HID + 255)/256, 256, 0, stream>>>(
      xgat, mean1, rstd1, gat_bn_g, gat_bn_b, NTOT, HID);
  // 4. GIN pre-aggregate (async global->LDS gather)
  gin_pre_kernel<<<NTOT, 256, 0, stream>>>(xgat, knn, gin_eps, bufA);
  // 5. hgin = bufA @ gin_w1 + gin_b1         (WMMA, 16x64 tiles)
  {
    int ntN = HID/64, tot = ntN * (NTOT/16);
    gemm_wmma_kernel<<<(tot + 3)/4, 128, 0, stream>>>(
        bufA, w1t, gin_b1, nullptr, bufB, NTOT, HID, HID, 0, ntN, tot);
  }
  // 6. BN + ReLU (in place)
  bn_stats_kernel<<<HID, 256, 0, stream>>>(bufB, mean2, rstd2, NTOT, HID);
  bn_apply_relu_kernel<<<(NTOT*HID + 255)/256, 256, 0, stream>>>(
      bufB, mean2, rstd2, gin_bn_g, gin_bn_b, NTOT, HID);
  // 7. x_gin = bufB @ gin_w2 + gin_b2        (WMMA)
  {
    int ntN = OUTC/64, tot = ntN * (NTOT/16);
    gemm_wmma_kernel<<<(tot + 3)/4, 128, 0, stream>>>(
        bufB, w2t, gin_b2, nullptr, bufC, NTOT, OUTC, HID, 0, ntN, tot);
  }
  // 8. node = relu(x_gin + x_gat @ res_w + res_b)  (WMMA, residual fused, in place)
  {
    int ntN = OUTC/64, tot = ntN * (NTOT/16);
    gemm_wmma_kernel<<<(tot + 3)/4, 128, 0, stream>>>(
        xgat, rwt, res_b, bufC, bufC, NTOT, OUTC, HID, 1, ntN, tot);
  }
  // 9. global mean pool (writes 16-row zero-padded graph)
  pool_kernel<<<dim3(16, OUTC/256), 256, 0, stream>>>(bufC, graph);
  // 10. out = graph @ fin_w + fin_b          (WMMA, M=8, A padded to 16 rows)
  {
    int ntN = (PREDL*ENCIN)/64, tot = ntN * 1;
    gemm_wmma_kernel<<<(tot + 3)/4, 128, 0, stream>>>(
        graph, fwt, fin_b, nullptr, (float*)d_out, BB, PREDL*ENCIN, OUTC, 0, ntN, tot);
  }
}
